// GroupedQueryAttention_82703890251918
// MI455X (gfx1250) — matrix operand order repaired and checked by reading, not compiled, not run on hardware
//
#include <hip/hip_runtime.h>

// ---------------------------------------------------------------------------
// GQA attention block for MI455X (gfx1250, wave32, WMMA).
// fp32 inputs -> bf16; all GEMM-shaped math on v_wmma_f32_16x16x32_bf16 with
// f32 accumulation. Flash-style attention (online softmax). Tile staging uses
// CDNA5 async global->LDS DMA (ASYNCcnt) when the toolchain exposes it, with
// double-buffered GEMM k-loop; otherwise falls back to load+ds_store.
// ---------------------------------------------------------------------------

typedef unsigned short ushort_t;
typedef __attribute__((ext_vector_type(16))) __bf16       v16bf;
typedef __attribute__((ext_vector_type(8)))  float        v8f;
typedef __attribute__((ext_vector_type(8)))  unsigned int v8u;
typedef __attribute__((ext_vector_type(4)))  int          i32x4;

static constexpr int BATCH   = 2;
static constexpr int SEQ     = 2048;
static constexpr int DMODEL  = 2048;
static constexpr int NHEADS  = 32;
static constexpr int NKV     = 8;
static constexpr int HDIM    = 64;
static constexpr int ROWS    = BATCH * SEQ;
static constexpr float SCALE = 0.125f;

// --------------------- async global->LDS copy (guarded) --------------------

#if defined(__has_builtin)
#if __has_builtin(__builtin_amdgcn_global_load_async_to_lds_b128) && \
    __has_builtin(__builtin_amdgcn_s_wait_asynccnt)
#define HAVE_ASYNC_LDS 1
#endif
#endif
#ifndef HAVE_ASYNC_LDS
#define HAVE_ASYNC_LDS 0
#endif

#if HAVE_ASYNC_LDS
typedef __attribute__((address_space(1))) i32x4 g_i32x4;   // global int4
typedef __attribute__((address_space(3))) i32x4 l_i32x4;   // LDS int4
#endif

// Copy 16 bytes global -> LDS (per lane). Async path lands via ASYNCcnt.
static __device__ inline void copy16_g2l(ushort_t* lds_dst, const ushort_t* gsrc) {
#if HAVE_ASYNC_LDS
  __builtin_amdgcn_global_load_async_to_lds_b128(
      (g_i32x4*)(size_t)gsrc,
      (l_i32x4*)lds_dst,
      0, 0);
#else
  uint4 t;
  __builtin_memcpy(&t, gsrc, 16);
  __builtin_memcpy(lds_dst, &t, 16);
#endif
}

static __device__ inline void copy_join() {
#if HAVE_ASYNC_LDS
  __builtin_amdgcn_s_wait_asynccnt(0);   // -> s_wait_asynccnt 0
#endif
}

// ----------------------------- small helpers ------------------------------

static __device__ inline unsigned short f2bf(float x) {
  union { float f; unsigned int u; } v; v.f = x;
  unsigned int u = v.u;
  unsigned int r = u + 0x7fffu + ((u >> 16) & 1u);   // RNE
  return (unsigned short)(r >> 16);
}

static __device__ inline v8f v8f_zero() {
  v8f z;
#pragma unroll
  for (int i = 0; i < 8; ++i) z[i] = 0.0f;
  return z;
}

static __device__ inline v16bf mk_frag(uint4 lo, uint4 hi) {
  v8u r;
  r[0] = lo.x; r[1] = lo.y; r[2] = lo.z; r[3] = lo.w;
  r[4] = hi.x; r[5] = hi.y; r[6] = hi.z; r[7] = hi.w;
  return __builtin_bit_cast(v16bf, r);
}

// A fragment 16x32 bf16 from row-major [m][k] LDS tile (this code's k order, not the ISA table's):
// lane L: M=L&15, half=L>>4; VGPR0-7 K=half*16+0..15 (contiguous, matches B).
static __device__ inline v16bf load_frag_A(const ushort_t* base, int row,
                                           int stride, int kbase, int hf) {
  const ushort_t* p = base + row * stride + kbase;
  uint4 lo, hi;
  __builtin_memcpy(&lo, p + hf * 16, 16);
  __builtin_memcpy(&hi, p + hf * 16 + 8, 16);
  return mk_frag(lo, hi);
}

// B fragment 32x16 bf16 where LDS holds [n][k] row-major:
// lane L: N=L&15, half=L>>4; VGPR v K=half*16+2v,2v+1 (contiguous 32B).
static __device__ inline v16bf load_frag_B(const ushort_t* base, int row,
                                           int stride, int kbase, int hf) {
  const ushort_t* p = base + row * stride + kbase + hf * 16;
  uint4 lo, hi;
  __builtin_memcpy(&lo, p, 16);
  __builtin_memcpy(&hi, p + 8, 16);
  return mk_frag(lo, hi);
}

static __device__ inline v8f wmma_bf16(v16bf a, v16bf b, v8f c) {
  return __builtin_amdgcn_wmma_f32_16x16x32_bf16(
      false, a, false, b, (short)0, c, false, false);
}

// --------------------------- conversion kernels ----------------------------

__global__ void __launch_bounds__(256)
cvt_f32_to_bf16(const float* __restrict__ src, ushort_t* __restrict__ dst, int n) {
  int i = blockIdx.x * 256 + threadIdx.x;
  if (i < n) dst[i] = f2bf(src[i]);
}

// RoPE + repack: src f32 [ROWS][H*64] -> dst bf16 [B][H][S][64]
__global__ void __launch_bounds__(256)
rope_repack(const float* __restrict__ src, ushort_t* __restrict__ dst, int H) {
  int idx = blockIdx.x * 256 + threadIdx.x;
  int total = BATCH * H * SEQ * 32;
  if (idx >= total) return;
  int i = idx & 31;
  int t = idx >> 5;
  int s = t & (SEQ - 1);
  t >>= 11;
  int h = t % H;
  int b = t / H;
  float inv = __expf(-(float)(2 * i) * (1.0f / 64.0f) * 9.210340371976184f);
  float ang = (float)s * inv;
  float sn, cs;
  __sincosf(ang, &sn, &cs);
  const float* ps = src + (size_t)(b * SEQ + s) * (H * HDIM) + h * HDIM + 2 * i;
  float xr = ps[0], xi = ps[1];
  ushort_t* pd = dst + (((size_t)(b * H + h) * SEQ + s) * HDIM + 2 * i);
  pd[0] = f2bf(xr * cs - xi * sn);
  pd[1] = f2bf(xr * sn + xi * cs);
}

// V repack: src f32 [ROWS][512] -> dst bf16 [B][KV][S][64]
__global__ void __launch_bounds__(256)
repack_v(const float* __restrict__ src, ushort_t* __restrict__ dst) {
  int idx = blockIdx.x * 256 + threadIdx.x;
  int total = BATCH * NKV * SEQ * HDIM;
  if (idx >= total) return;
  int d = idx & 63;
  int t = idx >> 6;
  int s = t & (SEQ - 1);
  t >>= 11;
  int kvh = t & 7;
  int b = t >> 3;
  dst[idx] = f2bf(src[(size_t)(b * SEQ + s) * (NKV * HDIM) + kvh * HDIM + d]);
}

// ------------------------------- GEMM (WMMA) -------------------------------
// C[M][N] f32 = A[M][K] bf16 * B[N][K]^T bf16. Tile 128x128x32, 8 waves.
// Double-buffered LDS; staging via async DMA when available. B fragments are
// batch-loaded before the 8-WMMA burst so the matrix pipe runs back-to-back.
__global__ void __launch_bounds__(256)
gemm_bf16_nt(const ushort_t* __restrict__ A, const ushort_t* __restrict__ B,
             float* __restrict__ C, int M, int N, int K) {
  __shared__ __align__(16) ushort_t As[2][128 * 32];   // 2 x 8 KB
  __shared__ __align__(16) ushort_t Bs[2][128 * 32];   // 2 x 8 KB

  const int tid  = threadIdx.x;
  const int w    = tid >> 5;
  const int lane = tid & 31;
  const int hf   = lane >> 4;
  const int l15  = lane & 15;
  const int m0   = blockIdx.y * 128;
  const int n0   = blockIdx.x * 128;

  // Per-thread staging coordinates (2 x 16B chunks per matrix).
  const int c0 = tid, c1 = tid + 256;
  const int r0 = c0 >> 2, r1 = c1 >> 2;
  const int x0 = (c0 & 3) * 8, x1 = (c1 & 3) * 8;

  v8f acc[8];
#pragma unroll
  for (int nt = 0; nt < 8; ++nt) acc[nt] = v8f_zero();

  // Prologue: stage k-tile 0 into buffer 0.
  copy16_g2l(&As[0][r0 * 32 + x0], A + (size_t)(m0 + r0) * K + x0);
  copy16_g2l(&As[0][r1 * 32 + x1], A + (size_t)(m0 + r1) * K + x1);
  copy16_g2l(&Bs[0][r0 * 32 + x0], B + (size_t)(n0 + r0) * K + x0);
  copy16_g2l(&Bs[0][r1 * 32 + x1], B + (size_t)(n0 + r1) * K + x1);
  copy_join();
  __syncthreads();

  for (int k0 = 0; k0 < K; k0 += 32) {
    const int buf = (k0 >> 5) & 1;
    // Stage next k-tile into the other buffer while computing this one.
    if (k0 + 32 < K) {
      const int nb = buf ^ 1;
      const int kn = k0 + 32;
      copy16_g2l(&As[nb][r0 * 32 + x0], A + (size_t)(m0 + r0) * K + kn + x0);
      copy16_g2l(&As[nb][r1 * 32 + x1], A + (size_t)(m0 + r1) * K + kn + x1);
      copy16_g2l(&Bs[nb][r0 * 32 + x0], B + (size_t)(n0 + r0) * K + kn + x0);
      copy16_g2l(&Bs[nb][r1 * 32 + x1], B + (size_t)(n0 + r1) * K + kn + x1);
    }

    v16bf af = load_frag_A(As[buf], 16 * w + l15, 32, 0, hf);
    v16bf bfr[8];
#pragma unroll
    for (int nt = 0; nt < 8; ++nt)
      bfr[nt] = load_frag_B(Bs[buf], 16 * nt + l15, 32, 0, hf);
#pragma unroll
    for (int nt = 0; nt < 8; ++nt)
      acc[nt] = wmma_bf16(af, bfr[nt], acc[nt]);

    copy_join();          // our async copies into the other buffer landed
    __syncthreads();      // everyone done reading buf & writing buf^1
  }

#pragma unroll
  for (int nt = 0; nt < 8; ++nt) {
#pragma unroll
    for (int r = 0; r < 8; ++r) {
      int row = m0 + 16 * w + 8 * hf + r;
      int col = n0 + 16 * nt + l15;
      C[(size_t)row * N + col] = acc[nt][r];
    }
  }
}

// --------------------------- flash attention -------------------------------
// Q bf16 [B][H][S][64], K/V bf16 [B][KV][S][64], O bf16 [B*S][DMODEL].
// Workgroup: 128 queries of one (b,h); 8 waves x 16 rows; 64-key blocks with
// online softmax; causal early exit.
__global__ void __launch_bounds__(256)
attn_flash(const ushort_t* __restrict__ Q, const ushort_t* __restrict__ Kmat,
           const ushort_t* __restrict__ V, ushort_t* __restrict__ O) {
  __shared__ __align__(16) ushort_t Qs[128 * 64];    // 16 KB
  __shared__ __align__(16) ushort_t Ks[64 * 64];     //  8 KB  [key][d]
  __shared__ __align__(16) ushort_t Vt[64 * 64];     //  8 KB  [d][key]
  __shared__ __align__(16) ushort_t Ps[8 * 16 * 64]; // 16 KB  per-wave P stage

  const int tid  = threadIdx.x;
  const int w    = tid >> 5;
  const int lane = tid & 31;
  const int hf   = lane >> 4;
  const int l15  = lane & 15;

  const int bh  = blockIdx.y;
  const int b   = bh >> 5;
  const int h   = bh & 31;
  const int kvh = h >> 2;                            // N_REP = 4
  const int q0  = blockIdx.x * 128;

  const ushort_t* qbase = Q + ((size_t)bh * SEQ + q0) * HDIM;
  const ushort_t* kbase = Kmat + (size_t)(b * NKV + kvh) * SEQ * HDIM;
  const ushort_t* vbase = V    + (size_t)(b * NKV + kvh) * SEQ * HDIM;

  // Stage the 128x64 Q tile (async DMA when available).
#pragma unroll
  for (int i = 0; i < 4; ++i) {
    int c = tid + i * 256;
    int row = c >> 3;
    int col = (c & 7) * 8;
    copy16_g2l(Qs + row * 64 + col, qbase + (size_t)row * HDIM + col);
  }

  v8f ctx[4];
#pragma unroll
  for (int a4 = 0; a4 < 4; ++a4) ctx[a4] = v8f_zero();
  float mrow[8], lrow[8];
#pragma unroll
  for (int r = 0; r < 8; ++r) { mrow[r] = -3.0e38f; lrow[r] = 0.0f; }

  ushort_t* myP = Ps + w * 16 * 64;
  const int nkb = q0 / 64 + 2;                       // causal block count

  copy_join();
  __syncthreads();

  for (int kb = 0; kb < nkb; ++kb) {
    __syncthreads();
    // K tile: straight copy (async path). V tile: VGPR round-trip transpose.
#pragma unroll
    for (int i = 0; i < 2; ++i) {
      int c = tid + i * 256;
      int row = c >> 3;
      int col = (c & 7) * 8;
      copy16_g2l(Ks + row * 64 + col,
                 kbase + (size_t)(kb * 64 + row) * HDIM + col);
      ushort_t vtmp[8];
      __builtin_memcpy(vtmp, vbase + (size_t)(kb * 64 + row) * HDIM + col, 16);
#pragma unroll
      for (int j = 0; j < 8; ++j) Vt[(col + j) * 64 + row] = vtmp[j];
    }
    copy_join();
    __syncthreads();
    if (kb + 1 < nkb) {                              // -> global_prefetch_b8
      __builtin_prefetch(kbase + (size_t)((kb + 1) * 64) * HDIM + tid * 16, 0, 1);
      __builtin_prefetch(vbase + (size_t)((kb + 1) * 64) * HDIM + tid * 16, 0, 1);
    }

    // S = Q * K^T : batch the fragment loads, then burst the WMMAs.
    v8f sc[4];
#pragma unroll
    for (int a4 = 0; a4 < 4; ++a4) sc[a4] = v8f_zero();
#pragma unroll
    for (int kk = 0; kk < 2; ++kk) {
      v16bf af = load_frag_A(Qs, 16 * w + l15, 64, 32 * kk, hf);
      v16bf bk[4];
#pragma unroll
      for (int a4 = 0; a4 < 4; ++a4)
        bk[a4] = load_frag_B(Ks, 16 * a4 + l15, 64, 32 * kk, hf);
#pragma unroll
      for (int a4 = 0; a4 < 4; ++a4)
        sc[a4] = wmma_bf16(af, bk[a4], sc[a4]);
    }

    // Scale + causal mask (mask element (q,k) iff k > q).
    const bool needmask = (kb * 64 + 63) > (q0 + 16 * w);
#pragma unroll
    for (int a4 = 0; a4 < 4; ++a4) {
#pragma unroll
      for (int r = 0; r < 8; ++r) {
        float v = sc[a4][r] * SCALE;
        if (needmask) {
          int qg = q0 + 16 * w + 8 * hf + r;
          int kg = kb * 64 + 16 * a4 + l15;
          if (kg > qg) v = -1.0e30f;
        }
        sc[a4][r] = v;
      }
    }

    // Online softmax per row (row = 16 lanes of one half-wave).
#pragma unroll
    for (int r = 0; r < 8; ++r) {
      float t = fmaxf(fmaxf(sc[0][r], sc[1][r]), fmaxf(sc[2][r], sc[3][r]));
#pragma unroll
      for (int msk = 1; msk < 16; msk <<= 1) t = fmaxf(t, __shfl_xor(t, msk, 32));
      float mnew  = fmaxf(mrow[r], t);
      float alpha = __expf(mrow[r] - mnew);
      mrow[r] = mnew;
      float rs = 0.0f;
#pragma unroll
      for (int a4 = 0; a4 < 4; ++a4) {
        float p = __expf(sc[a4][r] - mnew);
        myP[(8 * hf + r) * 64 + 16 * a4 + l15] = f2bf(p);  // C-layout -> LDS
        rs += p;
      }
#pragma unroll
      for (int msk = 1; msk < 16; msk <<= 1) rs += __shfl_xor(rs, msk, 32);
      lrow[r] = lrow[r] * alpha + rs;
#pragma unroll
      for (int a4 = 0; a4 < 4; ++a4) ctx[a4][r] *= alpha;
    }

    // ctx += P * V (P re-read in A-layout from this wave's private LDS).
#pragma unroll
    for (int kk = 0; kk < 2; ++kk) {
      v16bf af = load_frag_A(myP, l15, 64, 32 * kk, hf);
      v16bf bv[4];
#pragma unroll
      for (int a4 = 0; a4 < 4; ++a4)
        bv[a4] = load_frag_B(Vt, 16 * a4 + l15, 64, 32 * kk, hf);
#pragma unroll
      for (int a4 = 0; a4 < 4; ++a4)
        ctx[a4] = wmma_bf16(af, bv[a4], ctx[a4]);
    }
  }

  // Normalize, write context bf16 to [b*S + s][h*64 + d].
#pragma unroll
  for (int a4 = 0; a4 < 4; ++a4) {
#pragma unroll
    for (int r = 0; r < 8; ++r) {
      int row = q0 + 16 * w + 8 * hf + r;
      int col = h * HDIM + 16 * a4 + l15;
      O[(size_t)(b * SEQ + row) * DMODEL + col] = f2bf(ctx[a4][r] / lrow[r]);
    }
  }
}

// ------------------------------- launcher ----------------------------------

extern "C" void kernel_launch(void* const* d_in, const int* in_sizes, int n_in,
                              void* d_out, int out_size, void* d_ws, size_t ws_size,
                              hipStream_t stream) {
  (void)in_sizes; (void)n_in; (void)out_size; (void)ws_size;
  const float* x  = (const float*)d_in[0];
  const float* wq = (const float*)d_in[1];
  const float* wk = (const float*)d_in[2];
  const float* wv = (const float*)d_in[3];
  const float* wo = (const float*)d_in[4];
  float* out = (float*)d_out;

  const size_t xe  = (size_t)ROWS * DMODEL;
  const size_t wqe = (size_t)DMODEL * DMODEL;
  const size_t wke = (size_t)(NKV * HDIM) * DMODEL;
  const size_t qfe = (size_t)ROWS * DMODEL;
  const size_t kfe = (size_t)ROWS * (NKV * HDIM);

  char* p = (char*)d_ws;
  ushort_t* x_bf  = (ushort_t*)p; p += xe  * 2;
  ushort_t* wq_bf = (ushort_t*)p; p += wqe * 2;
  ushort_t* wk_bf = (ushort_t*)p; p += wke * 2;
  ushort_t* wv_bf = (ushort_t*)p; p += wke * 2;
  ushort_t* wo_bf = (ushort_t*)p; p += wqe * 2;
  float*    q_f   = (float*)p;    p += qfe * 4;
  float*    k_f   = (float*)p;    p += kfe * 4;
  float*    v_f   = (float*)p;    p += kfe * 4;
  ushort_t* q_bf  = (ushort_t*)p; p += qfe * 2;
  ushort_t* k_bf  = (ushort_t*)p; p += kfe * 2;
  ushort_t* v_bf  = (ushort_t*)p; p += kfe * 2;
  ushort_t* ctx_bf= (ushort_t*)p; p += xe  * 2;

  cvt_f32_to_bf16<<<(xe  + 255) / 256, 256, 0, stream>>>(x,  x_bf,  (int)xe);
  cvt_f32_to_bf16<<<(wqe + 255) / 256, 256, 0, stream>>>(wq, wq_bf, (int)wqe);
  cvt_f32_to_bf16<<<(wke + 255) / 256, 256, 0, stream>>>(wk, wk_bf, (int)wke);
  cvt_f32_to_bf16<<<(wke + 255) / 256, 256, 0, stream>>>(wv, wv_bf, (int)wke);
  cvt_f32_to_bf16<<<(wqe + 255) / 256, 256, 0, stream>>>(wo, wo_bf, (int)wqe);

  gemm_bf16_nt<<<dim3(DMODEL / 128, ROWS / 128), 256, 0, stream>>>(
      x_bf, wq_bf, q_f, ROWS, DMODEL, DMODEL);
  gemm_bf16_nt<<<dim3((NKV * HDIM) / 128, ROWS / 128), 256, 0, stream>>>(
      x_bf, wk_bf, k_f, ROWS, NKV * HDIM, DMODEL);
  gemm_bf16_nt<<<dim3((NKV * HDIM) / 128, ROWS / 128), 256, 0, stream>>>(
      x_bf, wv_bf, v_f, ROWS, NKV * HDIM, DMODEL);

  {
    int tq = BATCH * NHEADS * SEQ * 32;
    rope_repack<<<(tq + 255) / 256, 256, 0, stream>>>(q_f, q_bf, NHEADS);
    int tk = BATCH * NKV * SEQ * 32;
    rope_repack<<<(tk + 255) / 256, 256, 0, stream>>>(k_f, k_bf, NKV);
    int tv = BATCH * NKV * SEQ * HDIM;
    repack_v<<<(tv + 255) / 256, 256, 0, stream>>>(v_f, v_bf);
  }

  attn_flash<<<dim3(SEQ / 128, BATCH * NHEADS), 256, 0, stream>>>(
      q_bf, k_bf, v_bf, ctx_bf);

  gemm_bf16_nt<<<dim3(DMODEL / 128, ROWS / 128), 256, 0, stream>>>(
      ctx_bf, wo_bf, out, ROWS, DMODEL, DMODEL);
}
